// VesselSegmVGN_26456998543491
// MI455X (gfx1250) — matrix operand order repaired and checked by reading, not compiled, not run on hardware
//
#include <hip/hip_runtime.h>
#include <math.h>

// ---------------------------------------------------------------------------
// Types for WMMA bf16 16x16x32 (CDNA5 / gfx1250, wave32)
// bf16 is stored in memory as unsigned short (raw bits).
// ---------------------------------------------------------------------------
typedef __bf16 bf16_t;
typedef bf16_t v16bf __attribute__((ext_vector_type(16)));
typedef float  v8f   __attribute__((ext_vector_type(8)));

union ABf {
  v16bf v;
  unsigned short u[16];
  unsigned int w[8];
  uint4 q[2];
};

__device__ __forceinline__ unsigned short f2bf(float f) {
  unsigned int u = __float_as_uint(f);
  u += 0x7FFFu + ((u >> 16) & 1u);   // round-to-nearest-even
  return (unsigned short)(u >> 16);
}
__device__ __forceinline__ float bf2f(unsigned short u) {
  return __uint_as_float((unsigned int)u << 16);
}

// ---------------------------------------------------------------------------
// Utility kernels
// ---------------------------------------------------------------------------
__global__ void k_fill(float* p, float v, int n) {
  int i = blockIdx.x * blockDim.x + threadIdx.x;
  if (i < n) p[i] = v;
}

__global__ void k_f2bf(const float* __restrict__ in, unsigned short* __restrict__ out, long n) {
  long i = (long)blockIdx.x * blockDim.x + threadIdx.x;
  if (i < n) out[i] = f2bf(in[i]);
}

// Pack fp32 HWIO weights [KH,KW,Cin,Cout] into bf16 WMMA-B tiles.
// packed[((nt*KC + kc)*32 + lane)*16 + j]: lane L holds column N = nt*16+L%16,
// K = kc*32 + (L/16)*16 + j  (16 contiguous K per lane -> one 32B load).
__global__ void k_pack_w(const float* __restrict__ w, unsigned short* __restrict__ out,
                         int KH, int KW, int Cin, int Cout, int KC) {
  long idx = (long)blockIdx.x * blockDim.x + threadIdx.x;
  long total = (long)(Cout >> 4) * KC * 512;
  if (idx >= total) return;
  int j = (int)(idx & 15);
  int lane = (int)((idx >> 4) & 31);
  long r = idx >> 9;
  int kc = (int)(r % KC);
  int nt = (int)(r / KC);
  int Ktot = KH * KW * Cin;
  int n = nt * 16 + (lane & 15);
  int K = kc * 32 + (lane >> 4) * 16 + j;
  float val = 0.f;
  if (K < Ktot) {
    int t = K / Cin, c = K - t * Cin;
    val = w[((long)t * Cin + c) * Cout + n];
  }
  out[idx] = f2bf(val);
}

// ---------------------------------------------------------------------------
// Channel-aligned implicit-GEMM conv (Cin % 32 == 0), W power of two.
// bf16 activations in memory: A fragment = two global_load_b128 per lane.
// One wave -> 16 pixels x (16*NT) out channels; NT wmma per A fragment.
// Doubles as dense GEMM with KH=KW=1, H=1, W=M.
// ---------------------------------------------------------------------------
template <int NT>
__global__ __launch_bounds__(32)
void k_conv_al(const unsigned short* __restrict__ in, const unsigned short* __restrict__ wp,
               const float* __restrict__ bias, unsigned short* __restrict__ outb,
               float* __restrict__ outf,
               int H, int W, int wsh, int Cin, int KH, int KW,
               int outCs, int outCo, int relu) {
  int lane = threadIdx.x;
  int mt = blockIdx.x, ntb = blockIdx.y * NT, b = blockIdx.z;
  int CC = Cin >> 5;               // 32-channel chunks per tap
  int KC = KH * KW * CC;
  int l16 = lane & 15, hi = lane >> 4;
  int px = mt * 16 + l16;          // A-matrix row M for this lane
  int y = px >> wsh, x = px & (W - 1);
  const unsigned short* inb = in + (long)b * H * W * Cin;

  const unsigned short* bptr[NT];
#pragma unroll
  for (int tt = 0; tt < NT; ++tt)
    bptr[tt] = wp + (long)(ntb + tt) * KC * 512 + (long)lane * 16;

  v8f acc[NT];
#pragma unroll
  for (int tt = 0; tt < NT; ++tt) acc[tt] = {0.f, 0.f, 0.f, 0.f, 0.f, 0.f, 0.f, 0.f};

  for (int ky = 0; ky < KH; ++ky) {
    for (int kx = 0; kx < KW; ++kx) {
      int iy = y + ky - (KH >> 1);
      int ix = x + kx - (KW >> 1);
      bool ok = (iy >= 0) & (iy < H) & (ix >= 0) & (ix < W);
      const unsigned short* rowp = inb + ((long)iy * W + ix) * Cin;
      for (int cc = 0; cc < CC; ++cc) {
        ABf a;
        if (ok) {
          // lane's A data: channels cc*32 + hi*16 + [0..15]
          a.q[0] = ((const uint4*)(rowp + cc * 32 + hi * 16))[0];
          a.q[1] = ((const uint4*)(rowp + cc * 32 + hi * 16 + 8))[0];
        } else {
#pragma unroll
          for (int k = 0; k < 8; ++k) a.w[k] = 0u;
        }
#pragma unroll
        for (int tt = 0; tt < NT; ++tt) {
          ABf bm;
          const uint4* bq = (const uint4*)bptr[tt];
          bm.q[0] = bq[0];
          bm.q[1] = bq[1];
          bptr[tt] += 512;
          acc[tt] = __builtin_amdgcn_wmma_f32_16x16x32_bf16(false, a.v, false, bm.v,
                                                            (short)0, acc[tt], false, false);
        }
      }
    }
  }

  unsigned short* obb = outb + (long)b * H * W * outCs;
  float* obf = outf ? outf + (long)b * H * W * outCs : nullptr;
#pragma unroll
  for (int tt = 0; tt < NT; ++tt) {
    int n = (ntb + tt) * 16 + l16;
    float bv = bias ? bias[n] : 0.f;
#pragma unroll
    for (int v = 0; v < 8; ++v) {
      int M = hi * 8 + v;          // C/D layout: lane holds column n, rows hi*8+v
      int p = mt * 16 + M;
      float r = acc[tt][v] + bv;
      if (relu) r = r > 0.f ? r : 0.f;
      long idx = (long)p * outCs + outCo + n;
      obb[idx] = f2bf(r);
      if (obf) obf[idx] = r;
    }
  }
}

// ---------------------------------------------------------------------------
// Generic implicit-GEMM conv (any Cin, zero-padded K) — used for conv1_1 only.
// ---------------------------------------------------------------------------
__global__ __launch_bounds__(32)
void k_conv_wmma(const unsigned short* __restrict__ in, const unsigned short* __restrict__ wp,
                 const float* __restrict__ bias, unsigned short* __restrict__ outb,
                 int H, int W, int wsh, int Cin, int Cout, int KH, int KW,
                 int outCs, int outCo, int relu) {
  int lane = threadIdx.x;
  int mt = blockIdx.x, nt = blockIdx.y, b = blockIdx.z;
  int Ktot = KH * KW * Cin;
  int KC = (Ktot + 31) >> 5;
  int l16 = lane & 15, hi = lane >> 4;
  int px = mt * 16 + l16;
  int y = px >> wsh, x = px & (W - 1);
  const unsigned short* inb = in + (long)b * H * W * Cin;
  const unsigned short* wb = wp + (long)nt * KC * 512;
  v8f acc = {0.f, 0.f, 0.f, 0.f, 0.f, 0.f, 0.f, 0.f};

  for (int kc = 0; kc < KC; ++kc) {
    ABf a, bm;
    const uint4* bq = (const uint4*)(wb + ((long)kc * 32 + lane) * 16);
    bm.q[0] = bq[0];
    bm.q[1] = bq[1];
#pragma unroll
    for (int j = 0; j < 16; ++j) {
      int Kl = hi * 16 + j;
      int K = kc * 32 + Kl;
      unsigned short v = 0;
      if (K < Ktot) {
        int t = K / Cin, c = K - t * Cin;
        int ky = t / KW - (KH >> 1), kx = t - (t / KW) * KW - (KW >> 1);
        int iy = y + ky, ix = x + kx;
        if (iy >= 0 && iy < H && ix >= 0 && ix < W)
          v = inb[((long)iy * W + ix) * Cin + c];
      }
      a.u[j] = v;
    }
    acc = __builtin_amdgcn_wmma_f32_16x16x32_bf16(false, a.v, false, bm.v,
                                                  (short)0, acc, false, false);
  }

  int n = nt * 16 + l16;
  float bv = bias ? bias[n] : 0.f;
  unsigned short* ob = outb + (long)b * H * W * outCs;
#pragma unroll
  for (int v = 0; v < 8; ++v) {
    int M = hi * 8 + v;
    int p = mt * 16 + M;
    float r = acc[v] + bv;
    if (relu) r = r > 0.f ? r : 0.f;
    ob[(long)p * outCs + outCo + n] = f2bf(r);
  }
}

// ---------------------------------------------------------------------------
// 2x2 max pool, NHWC, bf16
// ---------------------------------------------------------------------------
__global__ void k_pool(const unsigned short* __restrict__ in, unsigned short* __restrict__ out,
                       int Bn, int H, int W, int C) {
  int Ho = H >> 1, Wo = W >> 1;
  long i = (long)blockIdx.x * blockDim.x + threadIdx.x;
  long total = (long)Bn * Ho * Wo * C;
  if (i >= total) return;
  int c = (int)(i % C);
  long r = i / C;
  int x = (int)(r % Wo); r /= Wo;
  int y = (int)(r % Ho); int b = (int)(r / Ho);
  const unsigned short* p = in + (((long)b * H + 2 * y) * W + 2 * x) * C + c;
  float m0 = bf2f(p[0]), m1 = bf2f(p[C]);
  float m2 = bf2f(p[(long)W * C]), m3 = bf2f(p[(long)W * C + C]);
  out[i] = f2bf(fmaxf(fmaxf(m0, m1), fmaxf(m2, m3)));
}

// ---------------------------------------------------------------------------
// Bilinear conv_transpose (diagonal HWIO kernel) + ReLU, bf16 in/out slice
// ---------------------------------------------------------------------------
__global__ void k_upsample(const unsigned short* __restrict__ in, const float* __restrict__ w,
                           unsigned short* __restrict__ out,
                           int Bn, int Hi, int Wi, int C, int s, int K,
                           int outCs, int outCo) {
  int Ho = Hi * s, Wo = Wi * s;
  long i = (long)blockIdx.x * blockDim.x + threadIdx.x;
  long total = (long)Bn * Ho * Wo * C;
  if (i >= total) return;
  int c = (int)(i % C);
  long r = i / C;
  int ox = (int)(r % Wo); r /= Wo;
  int oy = (int)(r % Ho); int b = (int)(r / Ho);
  int pb = (K - s) >> 1;
  float acc = 0.f;
  for (int ky = 0; ky < K; ++ky) {
    int ty = oy + pb - ky;
    if (ty < 0 || (ty % s) != 0) continue;
    int iy = ty / s;
    if (iy >= Hi) continue;
    for (int kx = 0; kx < K; ++kx) {
      int tx = ox + pb - kx;
      if (tx < 0 || (tx % s) != 0) continue;
      int ix = tx / s;
      if (ix >= Wi) continue;
      float wk = w[(((long)ky * K + kx) * C + c) * C + c];
      acc += wk * bf2f(in[(((long)b * Hi + iy) * Wi + ix) * C + c]);
    }
  }
  acc = acc > 0.f ? acc : 0.f;
  out[(((long)b * Ho + oy) * Wo + ox) * outCs + outCo + c] = f2bf(acc);
}

// ---------------------------------------------------------------------------
// 1x1 conv 64->1 (img logits), bf16 feat -> fp32 out
// ---------------------------------------------------------------------------
__global__ void k_out1x1(const unsigned short* __restrict__ feat, const float* __restrict__ w,
                         const float* __restrict__ b, float* __restrict__ out, long npix) {
  long i = (long)blockIdx.x * blockDim.x + threadIdx.x;
  if (i >= npix) return;
  const unsigned short* f = feat + i * 64;
  float acc = b[0];
#pragma unroll
  for (int c = 0; c < 64; ++c) acc += bf2f(f[c]) * w[c];
  out[i] = acc;
}

// Direct 3x3 conv, Cout=1, no activation (post logits), bf16 in
__global__ void k_conv_c1(const unsigned short* __restrict__ in, const float* __restrict__ w,
                          const float* __restrict__ b, float* __restrict__ out,
                          int Bn, int H, int W, int Cin) {
  long i = (long)blockIdx.x * blockDim.x + threadIdx.x;
  long total = (long)Bn * H * W;
  if (i >= total) return;
  int x = (int)(i % W);
  long r = i / W;
  int y = (int)(r % H); int bb = (int)(r / H);
  float acc = b[0];
  for (int ky = -1; ky <= 1; ++ky) {
    int iy = y + ky;
    if (iy < 0 || iy >= H) continue;
    for (int kx = -1; kx <= 1; ++kx) {
      int ix = x + kx;
      if (ix < 0 || ix >= W) continue;
      int t = (ky + 1) * 3 + (kx + 1);
      const unsigned short* ip = in + (((long)bb * H + iy) * W + ix) * Cin;
      const float* wp = w + (long)t * Cin;
      for (int c = 0; c < Cin; ++c) acc += bf2f(ip[c]) * wp[c];
    }
  }
  out[i] = acc;
}

// ---------------------------------------------------------------------------
// GNN kernels
// ---------------------------------------------------------------------------
__global__ void k_gather_nodes(const unsigned short* __restrict__ feat,
                               unsigned short* __restrict__ nf) {
  int i = blockIdx.x * blockDim.x + threadIdx.x;   // 2048*64
  if (i >= 2048 * 64) return;
  int c = i & 63, n = i >> 6;
  int b = n >> 10, rem = n & 1023;
  int gy = rem >> 5, gx = rem & 31;
  int y = gy * 8 + 4, x = gx * 8 + 4;
  nf[i] = feat[(((long)b * 256 + y) * 256 + x) * 64 + c];
}

__global__ void k_gat_scores(const float* __restrict__ h, const float* __restrict__ asrc,
                             const float* __restrict__ adst, float* __restrict__ es,
                             float* __restrict__ ed, int Nn, int heads, int F) {
  int i = blockIdx.x * blockDim.x + threadIdx.x;
  if (i >= Nn * heads) return;
  int hh = i % heads, n = i / heads;
  const float* hp = h + ((long)n * heads + hh) * F;
  float s = 0.f, d = 0.f;
  for (int f = 0; f < F; ++f) {
    float v = hp[f];
    s += v * asrc[hh * F + f];
    d += v * adst[hh * F + f];
  }
  es[i] = s;
  ed[i] = d;
}

__device__ __forceinline__ void atomicMaxF(float* addr, float val) {
  unsigned int* ua = (unsigned int*)addr;
  unsigned int old = *ua;
  while (__uint_as_float(old) < val) {
    unsigned int assumed = old;
    old = atomicCAS(ua, assumed, __float_as_uint(val));
    if (old == assumed) break;
  }
}

__global__ void k_gat_e1(const int* __restrict__ edges, int E, int Nn,
                         const float* __restrict__ es, const float* __restrict__ ed,
                         float* __restrict__ ebuf, float* __restrict__ m, int heads) {
  int e = blockIdx.x * blockDim.x + threadIdx.x;
  if (e >= E + Nn) return;
  int s, d;
  if (e < E) { s = edges[2 * e]; d = edges[2 * e + 1]; } else { s = d = e - E; }
  for (int hh = 0; hh < heads; ++hh) {
    float v = es[s * heads + hh] + ed[d * heads + hh];
    v = v > 0.f ? v : 0.2f * v;   // leaky_relu(0.2)
    ebuf[(long)e * heads + hh] = v;
    atomicMaxF(&m[d * heads + hh], v);
  }
}

__global__ void k_gat_e2(const int* __restrict__ edges, int E, int Nn,
                         float* __restrict__ ebuf, const float* __restrict__ m,
                         float* __restrict__ den, int heads) {
  int e = blockIdx.x * blockDim.x + threadIdx.x;
  if (e >= E + Nn) return;
  int d;
  if (e < E) d = edges[2 * e + 1]; else d = e - E;
  for (int hh = 0; hh < heads; ++hh) {
    float ex = expf(ebuf[(long)e * heads + hh] - m[d * heads + hh]);
    ebuf[(long)e * heads + hh] = ex;
    atomicAdd(&den[d * heads + hh], ex);
  }
}

__global__ void k_gat_e3(const int* __restrict__ edges, int E, int Nn,
                         const float* __restrict__ ebuf, const float* __restrict__ den,
                         const float* __restrict__ h, float* __restrict__ out,
                         int heads, int F) {
  int e = blockIdx.x * blockDim.x + threadIdx.x;
  if (e >= E + Nn) return;
  int s, d;
  if (e < E) { s = edges[2 * e]; d = edges[2 * e + 1]; } else { s = d = e - E; }
  for (int hh = 0; hh < heads; ++hh) {
    float coef = ebuf[(long)e * heads + hh] / (den[d * heads + hh] + 1e-9f);
    const float* hs = h + ((long)s * heads + hh) * F;
    float* od = out + ((long)d * heads + hh) * F;
    for (int f = 0; f < F; ++f) atomicAdd(&od[f], coef * hs[f]);
  }
}

// ELU in place (fp32); optional bf16 mirror for the next WMMA GEMM
__global__ void k_elu(float* p, unsigned short* mirror, int n) {
  int i = blockIdx.x * blockDim.x + threadIdx.x;
  if (i >= n) return;
  float x = p[i];
  x = x > 0.f ? x : (expf(x) - 1.f);
  p[i] = x;
  if (mirror) mirror[i] = f2bf(x);
}

__global__ void k_node_cls(const float* __restrict__ h2, const float* __restrict__ w,
                           const float* __restrict__ b, float* __restrict__ out, int Nn) {
  int n = blockIdx.x * blockDim.x + threadIdx.x;
  if (n >= Nn) return;
  float acc = b[0];
#pragma unroll
  for (int c = 0; c < 32; ++c) acc += h2[n * 32 + c] * w[c];
  out[n] = acc;
}

// cat[b,y,x,0..63]=feat(bf16) ; cat[b,y,x,64..95]=h2[node](fp32->bf16)
__global__ void k_build_cat(const unsigned short* __restrict__ feat, const float* __restrict__ h2,
                            unsigned short* __restrict__ cat) {
  long i = (long)blockIdx.x * blockDim.x + threadIdx.x;
  long total = (long)2 * 256 * 256 * 96;
  if (i >= total) return;
  int ch = (int)(i % 96);
  long p = i / 96;
  int x = (int)(p % 256);
  long r = p / 256;
  int y = (int)(r % 256); int b = (int)(r / 256);
  unsigned short v;
  if (ch < 64) {
    v = feat[p * 64 + ch];
  } else {
    int node = ((b * 32 + (y >> 3)) * 32 + (x >> 3));
    v = f2bf(h2[node * 32 + (ch - 64)]);
  }
  cat[i] = v;
}

// ---------------------------------------------------------------------------
// Host-side orchestration
// ---------------------------------------------------------------------------
extern "C" void kernel_launch(void* const* d_in, const int* in_sizes, int n_in,
                              void* d_out, int out_size, void* d_ws, size_t ws_size,
                              hipStream_t stream) {
  (void)in_sizes; (void)n_in; (void)out_size; (void)ws_size;
  const int Bn = 2, E = 65536, Nn = 2048, EE = E + Nn;
  typedef unsigned short u16;

  const float* imgs = (const float*)d_in[0];
  const int* edges = (const int*)d_in[1];
  const float* W_[47]; for (int i = 2; i < 47; ++i) W_[i] = (const float*)d_in[i];

  // bump allocator over d_ws
  char* base = (char*)d_ws;
  size_t off = 0;
  auto alloc = [&](size_t bytes) -> void* {
    void* p = base + off;
    off = (off + bytes + 255) & ~(size_t)255;
    return p;
  };
  auto allocf = [&](size_t elems) -> float* { return (float*)alloc(elems * 4); };
  auto allocb = [&](size_t elems) -> u16* { return (u16*)alloc(elems * 2); };

  auto fill = [&](float* p, float v, long n) {
    k_fill<<<dim3((unsigned)((n + 255) / 256)), 256, 0, stream>>>(p, v, (int)n);
  };
  auto pack = [&](const float* w, int KH, int KW, int Cin, int Cout) -> u16* {
    int KC = (KH * KW * Cin + 31) / 32;
    long total = (long)(Cout / 16) * KC * 512;
    u16* p = (u16*)alloc((size_t)total * 2);
    k_pack_w<<<dim3((unsigned)((total + 255) / 256)), 256, 0, stream>>>(w, p, KH, KW, Cin, Cout, KC);
    return p;
  };
  auto wshift = [](int W) { int s = 0; while ((1 << s) < W) ++s; return s; };
  auto conv = [&](const u16* in, const u16* wp, const float* bias,
                  u16* outb, float* outf,
                  int B2, int H, int W, int Cin, int Cout, int KH, int KW,
                  int outCs, int outCo, int relu) {
    int wsh = wshift(W);
    if ((Cin & 31) == 0) {
      if ((Cout & 63) == 0) {
        dim3 g((unsigned)(H * W / 16), (unsigned)(Cout / 64), (unsigned)B2);
        k_conv_al<4><<<g, 32, 0, stream>>>(in, wp, bias, outb, outf, H, W, wsh, Cin, KH, KW, outCs, outCo, relu);
      } else if ((Cout & 31) == 0) {
        dim3 g((unsigned)(H * W / 16), (unsigned)(Cout / 32), (unsigned)B2);
        k_conv_al<2><<<g, 32, 0, stream>>>(in, wp, bias, outb, outf, H, W, wsh, Cin, KH, KW, outCs, outCo, relu);
      } else {
        dim3 g((unsigned)(H * W / 16), (unsigned)(Cout / 16), (unsigned)B2);
        k_conv_al<1><<<g, 32, 0, stream>>>(in, wp, bias, outb, outf, H, W, wsh, Cin, KH, KW, outCs, outCo, relu);
      }
    } else {
      dim3 g((unsigned)(H * W / 16), (unsigned)(Cout / 16), (unsigned)B2);
      k_conv_wmma<<<g, 32, 0, stream>>>(in, wp, bias, outb, H, W, wsh, Cin, Cout, KH, KW, outCs, outCo, relu);
    }
  };

  // ---- pack all WMMA weights (bf16 B-layout) ----
  u16* pc11 = pack(W_[2], 3, 3, 3, 64);
  u16* pc12 = pack(W_[4], 3, 3, 64, 64);
  u16* pc21 = pack(W_[6], 3, 3, 64, 128);
  u16* pc22 = pack(W_[8], 3, 3, 128, 128);
  u16* pc31 = pack(W_[10], 3, 3, 128, 256);
  u16* pc32 = pack(W_[12], 3, 3, 256, 256);
  u16* pc33 = pack(W_[14], 3, 3, 256, 256);
  u16* pc41 = pack(W_[16], 3, 3, 256, 512);
  u16* pc42 = pack(W_[18], 3, 3, 512, 512);
  u16* pc43 = pack(W_[20], 3, 3, 512, 512);
  u16* ps1 = pack(W_[22], 3, 3, 64, 16);
  u16* ps2 = pack(W_[24], 3, 3, 128, 16);
  u16* ps3 = pack(W_[26], 3, 3, 256, 16);
  u16* ps4 = pack(W_[28], 3, 3, 512, 16);
  u16* pg1 = pack(W_[35], 1, 1, 64, 64);
  u16* pg2 = pack(W_[38], 1, 1, 64, 32);
  u16* pi1 = pack(W_[43], 3, 3, 96, 32);

  // ---- activation buffers (bf16 datapath) ----
  u16* imgs_bf = allocb((size_t)Bn * 256 * 256 * 3);
  u16* t1a = allocb((size_t)Bn * 256 * 256 * 64);
  u16* t1b = allocb((size_t)Bn * 256 * 256 * 64);
  u16* p1 = allocb((size_t)Bn * 128 * 128 * 64);
  u16* t2a = allocb((size_t)Bn * 128 * 128 * 128);
  u16* t2b = allocb((size_t)Bn * 128 * 128 * 128);
  u16* p2 = allocb((size_t)Bn * 64 * 64 * 128);
  u16* t3a = allocb((size_t)Bn * 64 * 64 * 256);
  u16* t3b = allocb((size_t)Bn * 64 * 64 * 256);
  u16* p3 = allocb((size_t)Bn * 32 * 32 * 256);
  u16* t4a = allocb((size_t)Bn * 32 * 32 * 512);
  u16* t4b = allocb((size_t)Bn * 32 * 32 * 512);
  u16* feat = allocb((size_t)Bn * 256 * 256 * 64);
  u16* stmp = allocb((size_t)Bn * 128 * 128 * 16);
  u16* nodef = allocb((size_t)Nn * 64);
  u16* hlin_bf = allocb((size_t)Nn * 64);
  u16* h1_bf = allocb((size_t)Nn * 64);
  u16* h2lin_bf = allocb((size_t)Nn * 32);
  u16* cat = allocb((size_t)Bn * 256 * 256 * 96);
  u16* post = allocb((size_t)Bn * 256 * 256 * 32);
  float* hlin = allocf((size_t)Nn * 64);
  float* agg1 = allocf((size_t)Nn * 64);
  float* h2lin = allocf((size_t)Nn * 32);
  float* agg2 = allocf((size_t)Nn * 32);
  float* esb = allocf((size_t)Nn * 4);
  float* edb = allocf((size_t)Nn * 4);
  float* mb = allocf((size_t)Nn * 4);
  float* den = allocf((size_t)Nn * 4);
  float* ebuf = allocf((size_t)EE * 4);

  // ---- convert input image to bf16 ----
  {
    long n = (long)Bn * 256 * 256 * 3;
    k_f2bf<<<dim3((unsigned)((n + 255) / 256)), 256, 0, stream>>>(imgs, imgs_bf, n);
  }

  // ---- backbone ----
  conv(imgs_bf, pc11, W_[3], t1a, nullptr, Bn, 256, 256, 3, 64, 3, 3, 64, 0, 1);
  conv(t1a, pc12, W_[5], t1b, nullptr, Bn, 256, 256, 64, 64, 3, 3, 64, 0, 1);   // c1
  k_pool<<<dim3((unsigned)(((long)Bn * 128 * 128 * 64 + 255) / 256)), 256, 0, stream>>>(t1b, p1, Bn, 256, 256, 64);
  conv(p1, pc21, W_[7], t2a, nullptr, Bn, 128, 128, 64, 128, 3, 3, 128, 0, 1);
  conv(t2a, pc22, W_[9], t2b, nullptr, Bn, 128, 128, 128, 128, 3, 3, 128, 0, 1); // c2
  k_pool<<<dim3((unsigned)(((long)Bn * 64 * 64 * 128 + 255) / 256)), 256, 0, stream>>>(t2b, p2, Bn, 128, 128, 128);
  conv(p2, pc31, W_[11], t3a, nullptr, Bn, 64, 64, 128, 256, 3, 3, 256, 0, 1);
  conv(t3a, pc32, W_[13], t3b, nullptr, Bn, 64, 64, 256, 256, 3, 3, 256, 0, 1);
  conv(t3b, pc33, W_[15], t3a, nullptr, Bn, 64, 64, 256, 256, 3, 3, 256, 0, 1);  // c3
  k_pool<<<dim3((unsigned)(((long)Bn * 32 * 32 * 256 + 255) / 256)), 256, 0, stream>>>(t3a, p3, Bn, 64, 64, 256);
  conv(p3, pc41, W_[17], t4a, nullptr, Bn, 32, 32, 256, 512, 3, 3, 512, 0, 1);
  conv(t4a, pc42, W_[19], t4b, nullptr, Bn, 32, 32, 512, 512, 3, 3, 512, 0, 1);
  conv(t4b, pc43, W_[21], t4a, nullptr, Bn, 32, 32, 512, 512, 3, 3, 512, 0, 1);  // c4

  // ---- SPE heads directly into feat channel slices ----
  conv(t1b, ps1, W_[23], feat, nullptr, Bn, 256, 256, 64, 16, 3, 3, 64, 0, 1);   // s1 -> ch 0..15
  conv(t2b, ps2, W_[25], stmp, nullptr, Bn, 128, 128, 128, 16, 3, 3, 16, 0, 1);
  k_upsample<<<dim3((unsigned)(((long)Bn * 256 * 256 * 16 + 255) / 256)), 256, 0, stream>>>(
      stmp, W_[30], feat, Bn, 128, 128, 16, 2, 4, 64, 16);
  conv(t3a, ps3, W_[27], stmp, nullptr, Bn, 64, 64, 256, 16, 3, 3, 16, 0, 1);
  k_upsample<<<dim3((unsigned)(((long)Bn * 256 * 256 * 16 + 255) / 256)), 256, 0, stream>>>(
      stmp, W_[31], feat, Bn, 64, 64, 16, 4, 8, 64, 32);
  conv(t4a, ps4, W_[29], stmp, nullptr, Bn, 32, 32, 512, 16, 3, 3, 16, 0, 1);
  k_upsample<<<dim3((unsigned)(((long)Bn * 256 * 256 * 16 + 255) / 256)), 256, 0, stream>>>(
      stmp, W_[32], feat, Bn, 32, 32, 16, 8, 16, 64, 48);

  // ---- img logits -> d_out[0 .. 131071] ----
  float* out_img = (float*)d_out;
  k_out1x1<<<dim3((unsigned)(((long)Bn * 256 * 256 + 255) / 256)), 256, 0, stream>>>(
      feat, W_[33], W_[34], out_img, (long)Bn * 256 * 256);

  // ---- GAT layer 1 (heads=4, F=16) ----
  k_gather_nodes<<<dim3((Nn * 64 + 255) / 256), 256, 0, stream>>>(feat, nodef);
  conv(nodef, pg1, nullptr, hlin_bf, hlin, 1, 1, Nn, 64, 64, 1, 1, 64, 0, 0);
  k_gat_scores<<<dim3((Nn * 4 + 255) / 256), 256, 0, stream>>>(hlin, W_[36], W_[37], esb, edb, Nn, 4, 16);
  fill(mb, -1e30f, Nn * 4);
  fill(den, 0.f, Nn * 4);
  fill(agg1, 0.f, Nn * 64);
  k_gat_e1<<<dim3((EE + 255) / 256), 256, 0, stream>>>(edges, E, Nn, esb, edb, ebuf, mb, 4);
  k_gat_e2<<<dim3((EE + 255) / 256), 256, 0, stream>>>(edges, E, Nn, ebuf, mb, den, 4);
  k_gat_e3<<<dim3((EE + 255) / 256), 256, 0, stream>>>(edges, E, Nn, ebuf, den, hlin, agg1, 4, 16);
  k_elu<<<dim3((Nn * 64 + 255) / 256), 256, 0, stream>>>(agg1, h1_bf, Nn * 64);  // h1 + bf16 mirror

  // ---- GAT layer 2 (heads=2, F=16) ----
  conv(h1_bf, pg2, nullptr, h2lin_bf, h2lin, 1, 1, Nn, 64, 32, 1, 1, 32, 0, 0);
  k_gat_scores<<<dim3((Nn * 2 + 255) / 256), 256, 0, stream>>>(h2lin, W_[39], W_[40], esb, edb, Nn, 2, 16);
  fill(mb, -1e30f, Nn * 2);
  fill(den, 0.f, Nn * 2);
  fill(agg2, 0.f, Nn * 32);
  k_gat_e1<<<dim3((EE + 255) / 256), 256, 0, stream>>>(edges, E, Nn, esb, edb, ebuf, mb, 2);
  k_gat_e2<<<dim3((EE + 255) / 256), 256, 0, stream>>>(edges, E, Nn, ebuf, mb, den, 2);
  k_gat_e3<<<dim3((EE + 255) / 256), 256, 0, stream>>>(edges, E, Nn, ebuf, den, h2lin, agg2, 2, 16);
  k_elu<<<dim3((Nn * 32 + 255) / 256), 256, 0, stream>>>(agg2, (u16*)nullptr, Nn * 32);  // h2

  // ---- node logits -> d_out[131072 .. 133119] ----
  float* out_node = (float*)d_out + (long)Bn * 256 * 256;
  k_node_cls<<<dim3((Nn + 255) / 256), 256, 0, stream>>>(agg2, W_[41], W_[42], out_node, Nn);

  // ---- inference module ----
  k_build_cat<<<dim3((unsigned)(((long)Bn * 256 * 256 * 96 + 255) / 256)), 256, 0, stream>>>(feat, agg2, cat);
  conv(cat, pi1, W_[44], post, nullptr, Bn, 256, 256, 96, 32, 3, 3, 32, 0, 1);
  float* out_post = out_node + Nn;
  k_conv_c1<<<dim3((unsigned)(((long)Bn * 256 * 256 + 255) / 256)), 256, 0, stream>>>(
      post, W_[45], W_[46], out_post, Bn, 256, 256, 32);
}
